// ResGCN_66915590472495
// MI455X (gfx1250) — matrix operand order repaired and checked by reading, not compiled, not run on hardware
//
#include <hip/hip_runtime.h>
#include <stdint.h>

#define FDIM 128

typedef __attribute__((ext_vector_type(16))) _Float16 v16h;
typedef __attribute__((ext_vector_type(8)))  float    v8f;

// ---------------- degree / norm precompute ----------------

__global__ void k_init_deg(float* deg, int n) {
    int i = blockIdx.x * blockDim.x + threadIdx.x;
    if (i < n) deg[i] = 1.0f;               // self-loop weight
}

__global__ void k_deg_accum(const int* __restrict__ dst, const float* __restrict__ ew,
                            float* deg, int e_cnt) {
    int e = blockIdx.x * blockDim.x + threadIdx.x;
    if (e < e_cnt) atomicAdd(&deg[dst[e]], ew[e]);
}

__global__ void k_dinv(float* deg, int n) {
    int i = blockIdx.x * blockDim.x + threadIdx.x;
    if (i < n) deg[i] = rsqrtf(deg[i]);     // in-place: deg -> dinv
}

__global__ void k_coef(const int* __restrict__ src, const int* __restrict__ dst,
                       const float* __restrict__ ew, const float* __restrict__ dinv,
                       float* __restrict__ coef, int e_cnt) {
    int e = blockIdx.x * blockDim.x + threadIdx.x;
    if (e < e_cnt) coef[e] = dinv[src[e]] * ew[e] * dinv[dst[e]];
}

// ---------------- weight swizzle into WMMA B-fragment layout ----------------
// frag idx = ((kc*8 + nt)*32 + lane)*16 + h
// B element: K = kc*32 + 8*(lane/16) + (h%8) + 16*(h/8) ; N = nt*16 + (lane%16)

__global__ void k_swizzle(const float* __restrict__ W, _Float16* __restrict__ out) {
    int idx = blockIdx.x * blockDim.x + threadIdx.x;
    if (idx >= 4 * 8 * 32 * 16) return;
    int h    = idx & 15;
    int lane = (idx >> 4) & 31;
    int nt   = (idx >> 9) & 7;
    int kc   = idx >> 12;
    int K    = kc * 32 + ((lane >> 4) << 3) + (h & 7) + ((h >> 3) << 4);
    int ncol = (nt << 4) + (lane & 15);
    out[idx] = (_Float16)W[K * FDIM + ncol];
}

// ---------------- WMMA GEMM: C[M,128] = A[M,128] @ W, f16 in / f32 acc ----------------
// B fragments staged once per block into LDS (32 KB), shared by all 8 waves.

__global__ __launch_bounds__(256) void k_gemm(const float* __restrict__ A,
                                              const _Float16* __restrict__ Bswz,
                                              float* __restrict__ C, int M) {
    __shared__ alignas(64) _Float16 Bs[4 * 8 * 32 * 16];   // 32 KB

    int tid  = threadIdx.x;
    int wave = blockIdx.x * (blockDim.x >> 5) + (tid >> 5);
    int lane = tid & 31;
    int row0 = wave << 4;                   // 16 rows per wave

    // cooperative stage of all B fragments: 32 KB = 2048 uint4, 8 per thread
    {
        const uint4* gsrc = (const uint4*)Bswz;
        uint4*       ldst = (uint4*)Bs;
#pragma unroll
        for (int i = 0; i < 8; ++i) ldst[tid + i * 256] = gsrc[tid + i * 256];
    }
    __syncthreads();

    if (row0 >= M) return;                  // wave-uniform: EXEC stays all-ones below

    int m  = lane & 15;                     // row within tile (A) / col within tile (B,C)
    int hs = lane >> 4;                     // lane half-select
    int o  = hs << 3;                       // A K-offset: 0 or 8

    const float* arow = A + (size_t)(row0 + m) * FDIM;

    v8f acc[8] = {};

#pragma unroll
    for (int kc = 0; kc < 4; ++kc) {
        // A fragment: halfs 0..7 -> K = kc*32 + o + {0..7}; halfs 8..15 -> +16
        const float4* q = (const float4*)(arow + kc * 32 + o);
        float4 f0 = q[0], f1 = q[1], f2 = q[4], f3 = q[5];
        v16h a;
        a[0]  = (_Float16)f0.x; a[1]  = (_Float16)f0.y; a[2]  = (_Float16)f0.z; a[3]  = (_Float16)f0.w;
        a[4]  = (_Float16)f1.x; a[5]  = (_Float16)f1.y; a[6]  = (_Float16)f1.z; a[7]  = (_Float16)f1.w;
        a[8]  = (_Float16)f2.x; a[9]  = (_Float16)f2.y; a[10] = (_Float16)f2.z; a[11] = (_Float16)f2.w;
        a[12] = (_Float16)f3.x; a[13] = (_Float16)f3.y; a[14] = (_Float16)f3.z; a[15] = (_Float16)f3.w;

#pragma unroll
        for (int nt = 0; nt < 8; ++nt) {
            v16h b = *(const v16h*)(Bs + (size_t)((kc * 8 + nt) * 32 + lane) * 16);
            acc[nt] = __builtin_amdgcn_wmma_f32_16x16x32_f16(
                false, a, false, b, (short)0, acc[nt], false, false);
        }
    }

    // D layout: VGPR r -> M = r + 8*hs, N = nt*16 + m
#pragma unroll
    for (int nt = 0; nt < 8; ++nt) {
        float* crow = C + (size_t)(row0 + (hs << 3)) * FDIM + (nt << 4) + m;
#pragma unroll
        for (int r = 0; r < 8; ++r) crow[(size_t)r * FDIM] = acc[nt][r];
    }
}

// ---------------- aggregation init: agg = t * dinv^2 (self loop) + bias ----------------

__global__ void k_agg_init(const float* __restrict__ t, const float* __restrict__ dinv,
                           const float* __restrict__ bias, float* __restrict__ agg, int n) {
    int idx = blockIdx.x * blockDim.x + threadIdx.x;   // float4 granularity: n*32 threads
    if (idx >= n * 32) return;
    int i  = idx >> 5;
    int f4 = idx & 31;
    float di = dinv[i];
    float s  = di * di;
    float4 tv = ((const float4*)t)[idx];
    float4 bv = ((const float4*)bias)[f4];
    float4 r;
    r.x = tv.x * s + bv.x; r.y = tv.y * s + bv.y;
    r.z = tv.z * s + bv.z; r.w = tv.w * s + bv.w;
    ((float4*)agg)[idx] = r;
}

// ---------------- edge scatter: agg[dst] += t[src] * coef, one wave per edge ----------------

__global__ void k_scatter(const float* __restrict__ t, const int* __restrict__ src,
                          const int* __restrict__ dst, const float* __restrict__ coef,
                          float* __restrict__ agg, int e_cnt) {
    int idx  = blockIdx.x * blockDim.x + threadIdx.x;
    int e    = idx >> 5;
    if (e >= e_cnt) return;
    int lane = idx & 31;
    int s = src[e];
    int d = dst[e];
    float c = coef[e];
    float4 v = *(const float4*)(t + (size_t)s * FDIM + lane * 4);
    float* out = agg + (size_t)d * FDIM + lane * 4;
    atomicAdd(out + 0, v.x * c);
    atomicAdd(out + 1, v.y * c);
    atomicAdd(out + 2, v.z * c);
    atomicAdd(out + 3, v.w * c);
}

// ---------------- combine: out = (agg + h), optional relu ----------------

__global__ void k_combine(const float* __restrict__ agg, const float* __restrict__ hbuf,
                          float* __restrict__ out, int total4, int do_relu) {
    int idx = blockIdx.x * blockDim.x + threadIdx.x;
    if (idx >= total4) return;
    float4 a = ((const float4*)agg)[idx];
    float4 b = ((const float4*)hbuf)[idx];
    float4 r;
    r.x = a.x + b.x; r.y = a.y + b.y; r.z = a.z + b.z; r.w = a.w + b.w;
    if (do_relu) {
        r.x = fmaxf(r.x, 0.f); r.y = fmaxf(r.y, 0.f);
        r.z = fmaxf(r.z, 0.f); r.w = fmaxf(r.w, 0.f);
    }
    ((float4*)out)[idx] = r;
}

// ---------------- driver ----------------

extern "C" void kernel_launch(void* const* d_in, const int* in_sizes, int n_in,
                              void* d_out, int out_size, void* d_ws, size_t ws_size,
                              hipStream_t stream) {
    const float* x      = (const float*)d_in[0];
    const int*   ei     = (const int*)  d_in[1];
    const float* ew     = (const float*)d_in[2];
    const float* W_init = (const float*)d_in[3];
    const float* W1     = (const float*)d_in[4];
    const float* b1     = (const float*)d_in[5];
    const float* W2     = (const float*)d_in[6];
    const float* b2     = (const float*)d_in[7];
    const float* W3     = (const float*)d_in[8];
    const float* b3     = (const float*)d_in[9];

    int N = in_sizes[0] / FDIM;
    int E = in_sizes[1] / 2;
    const int* src = ei;
    const int* dst = ei + E;

    float* out = (float*)d_out;             // also reused as per-layer GEMM output t
    size_t nd  = (size_t)N * FDIM;

    float* hbuf = (float*)d_ws;
    float* act  = hbuf + nd;
    float* agg  = act + nd;
    float* dinv = agg + nd;                 // deg -> dinv in place
    float* coef = dinv + N;
    uintptr_t p = (uintptr_t)(coef + E);
    p = (p + 255) & ~(uintptr_t)255;
    _Float16* wswz = (_Float16*)p;
    const int WFRAG = 4 * 8 * 32 * 16;      // 16384 halfs per matrix
    _Float16* wz0 = wswz;
    _Float16* wz1 = wswz + WFRAG;
    _Float16* wz2 = wswz + 2 * WFRAG;
    _Float16* wz3 = wswz + 3 * WFRAG;

    dim3 blk(256);
    int gN   = (N + 255) / 256;
    int gE   = (E + 255) / 256;
    int gNd4 = (int)((nd / 4 + 255) / 256);
    int gN32 = (int)(((size_t)N * 32 + 255) / 256);
    int gE32 = (int)(((size_t)E * 32 + 255) / 256);
    int gGem = (N + 127) / 128;             // 8 waves/block, 16 rows/wave

    // norms
    k_init_deg<<<gN, blk, 0, stream>>>(dinv, N);
    k_deg_accum<<<gE, blk, 0, stream>>>(dst, ew, dinv, E);
    k_dinv<<<gN, blk, 0, stream>>>(dinv, N);
    k_coef<<<gE, blk, 0, stream>>>(src, dst, ew, dinv, coef, E);

    // weight swizzles (f32 -> f16 B-fragments)
    k_swizzle<<<64, blk, 0, stream>>>(W_init, wz0);
    k_swizzle<<<64, blk, 0, stream>>>(W1, wz1);
    k_swizzle<<<64, blk, 0, stream>>>(W2, wz2);
    k_swizzle<<<64, blk, 0, stream>>>(W3, wz3);

    // h = x @ W_init
    k_gemm<<<gGem, blk, 0, stream>>>(x, wz0, hbuf, N);

    // layer 1: x1 = relu(conv(h, W1, b1) + h)
    k_gemm<<<gGem, blk, 0, stream>>>(hbuf, wz1, out, N);
    k_agg_init<<<gN32, blk, 0, stream>>>(out, dinv, b1, agg, N);
    k_scatter<<<gE32, blk, 0, stream>>>(out, src, dst, coef, agg, E);
    k_combine<<<gNd4, blk, 0, stream>>>(agg, hbuf, act, (int)(nd / 4), 1);

    // layer 2: x2 = relu(conv(x1, W2, b2) + h)
    k_gemm<<<gGem, blk, 0, stream>>>(act, wz2, out, N);
    k_agg_init<<<gN32, blk, 0, stream>>>(out, dinv, b2, agg, N);
    k_scatter<<<gE32, blk, 0, stream>>>(out, src, dst, coef, agg, E);
    k_combine<<<gNd4, blk, 0, stream>>>(agg, hbuf, act, (int)(nd / 4), 1);

    // layer 3: out = conv(x2, W3, b3) + h
    k_gemm<<<gGem, blk, 0, stream>>>(act, wz3, out, N);
    k_agg_init<<<gN32, blk, 0, stream>>>(out, dinv, b3, agg, N);
    k_scatter<<<gE32, blk, 0, stream>>>(out, src, dst, coef, agg, E);
    k_combine<<<gNd4, blk, 0, stream>>>(agg, hbuf, out, (int)(nd / 4), 0);
}
